// RNNLSTMTimeSeriesBinaryClassifier_74062416052959
// MI455X (gfx1250) — matrix operand order repaired and checked by reading, not compiled, not run on hardware
//
#include <hip/hip_runtime.h>
#include <math.h>

// ---------------------------------------------------------------------------
// RNN LSTM binary classifier for MI455X (gfx1250), wave32 + bf16 WMMA.
//
//  - x pre-converted to bf16 once (bandwidth-bound, ~17us at 23.3 TB/s) so
//    the projection GEMM's hot loop is pure b128 loads -> v_wmma.
//  - Projection GEMM: 2 M-tiles x NPW N-tiles per wave (B fragments reused
//    across 2 WMMAs -> halves L2 weight traffic), weights L2-resident.
//  - Recurrences sequential in t but independent across batch: 4 blocks x
//    16 batch rows, h (bf16) / c (f32) in LDS, Whh streamed from L2.
// ---------------------------------------------------------------------------

typedef __attribute__((ext_vector_type(16))) __bf16 v16bf;
typedef __attribute__((ext_vector_type(8)))  __bf16 v8bf;
typedef __attribute__((ext_vector_type(8)))  float  v8f;

constexpr int BATCH = 64;
constexpr int SEQ   = 512;
constexpr int IN1   = 2048;
constexpr int H1    = 256;
constexpr int H2    = 128;
constexpr int NFC1  = 64;

static __device__ __forceinline__ __bf16 f2bf(float f) {
    unsigned u = __builtin_bit_cast(unsigned, f);
    unsigned r = u + 0x7FFFu + ((u >> 16) & 1u);   // round-to-nearest-even
    unsigned short s = (unsigned short)(r >> 16);
    return __builtin_bit_cast(__bf16, s);
}
static __device__ __forceinline__ float bf2f(__bf16 b) {
    unsigned short s = __builtin_bit_cast(unsigned short, b);
    unsigned u = ((unsigned)s) << 16;
    return __builtin_bit_cast(float, u);
}
static __device__ __forceinline__ v16bf cat8(v8bf a, v8bf b) {
    return __builtin_shufflevector(a, b, 0,1,2,3,4,5,6,7,8,9,10,11,12,13,14,15);
}
static __device__ __forceinline__ float sigmoidf(float x) {
    return 1.0f / (1.0f + expf(-x));
}

// ------------------------- small prep kernels ------------------------------

__global__ void k_f2bf(const float* __restrict__ s, __bf16* __restrict__ d, int n) {
    for (int i = blockIdx.x * blockDim.x + threadIdx.x; i < n;
         i += gridDim.x * blockDim.x)
        d[i] = f2bf(s[i]);
}

__global__ void k_addbias(const float* __restrict__ a, const float* __restrict__ b,
                          float* __restrict__ o, int n) {
    for (int i = blockIdx.x * blockDim.x + threadIdx.x; i < n;
         i += gridDim.x * blockDim.x)
        o[i] = a[i] + b[i];
}

// ------------------------- batched input projection ------------------------
// Out[M][N] = A[M][K] * W[N][K]^T + bias[N]   (A, W bf16; Out f32)
// Block covers 2 M-tiles (32 rows) x full N; 8 waves, each owns NPW N-tiles.
// B fragment loaded once per (k, nt), reused for both M-tiles.

template<int NPW>
__global__ __launch_bounds__(256) void
k_xproj(const __bf16* __restrict__ A, const __bf16* __restrict__ W,
        const float* __restrict__ bias, float* __restrict__ Out,
        int K, int N) {
    const int tid  = threadIdx.x;
    const int wave = tid >> 5;
    const int lane = tid & 31;
    const int lm   = lane & 15;
    const int hi   = lane >> 4;
    const int nbase = wave * (NPW * 16);
    const long mrow0 = (long)blockIdx.x * 32 + lm;        // M-tile 0 row
    const long mrow1 = mrow0 + 16;                        // M-tile 1 row

    v8f acc[2][NPW];
#pragma unroll
    for (int mt = 0; mt < 2; ++mt)
#pragma unroll
        for (int nt = 0; nt < NPW; ++nt)
            acc[mt][nt] = v8f{0.f,0.f,0.f,0.f,0.f,0.f,0.f,0.f};

    for (int k0 = 0; k0 < K; k0 += 32) {
        // A operand: lane holds M=lm; K runs [k0+8hi..+7] and [k0+16+8hi..+7]
        const __bf16* Ar0 = A + mrow0 * K + k0 + 8 * hi;
        const __bf16* Ar1 = A + mrow1 * K + k0 + 8 * hi;
        v16bf a0 = cat8(*(const v8bf*)(Ar0), *(const v8bf*)(Ar0 + 16));
        v16bf a1 = cat8(*(const v8bf*)(Ar1), *(const v8bf*)(Ar1 + 16));
        // B operand: lane holds N=lm; K runs [k0+8hi..+7] and [k0+16+8hi..+7]
#pragma unroll
        for (int nt = 0; nt < NPW; ++nt) {
            const int n = nbase + nt * 16 + lm;
            const __bf16* Wrow = W + (long)n * K + k0 + 8 * hi;
            v16bf b = cat8(*(const v8bf*)(Wrow), *(const v8bf*)(Wrow + 16));
            acc[0][nt] = __builtin_amdgcn_wmma_f32_16x16x32_bf16(
                false, a0, false, b, (short)0, acc[0][nt], false, false);
            acc[1][nt] = __builtin_amdgcn_wmma_f32_16x16x32_bf16(
                false, a1, false, b, (short)0, acc[1][nt], false, false);
        }
    }
    // epilogue: C layout: VGPR r -> M=r+8*hi, col = nbase+nt*16+lm
#pragma unroll
    for (int mt = 0; mt < 2; ++mt) {
#pragma unroll
        for (int nt = 0; nt < NPW; ++nt) {
            const int ncol = nbase + nt * 16 + lm;
            const float bn = bias[ncol];
#pragma unroll
            for (int r = 0; r < 8; ++r) {
                const long orow = (long)blockIdx.x * 32 + mt * 16 + r + 8 * hi;
                Out[orow * N + ncol] = acc[mt][nt][r] + bn;
            }
        }
    }
}

// ------------------------- LSTM recurrence ---------------------------------
// Per block: 16 batch rows; sequential over t. gates = xg[t] + h @ Whh^T
// (M=16, K=H, N=4H bf16 WMMA), then activations + c/h update via LDS.

template<int H, bool WRITE_SEQ>
__global__ __launch_bounds__(256) void
k_lstm_rec(const float* __restrict__ xg, const __bf16* __restrict__ Whh,
           __bf16* __restrict__ outSeq, float* __restrict__ hLast, int T) {
    constexpr int N4     = 4 * H;
    constexpr int NPW    = N4 / (16 * 8);   // N-tiles per wave (8 waves)
    constexpr int KSTEPS = H / 32;

    __shared__ __attribute__((aligned(16))) __bf16 hL[16][H];       // h (A operand)
    __shared__ __attribute__((aligned(16))) float  cL[16][H];       // cell state
    __shared__ __attribute__((aligned(16))) __bf16 gAct[16][N4];    // activated gates

    const int tid  = threadIdx.x;
    const int wave = tid >> 5;
    const int lane = tid & 31;
    const int lm   = lane & 15;
    const int hi   = lane >> 4;
    const int nbase  = wave * (NPW * 16);
    const bool isTanh = ((wave >> 1) == 2);   // wave's columns all in one gate
    const int btile  = blockIdx.x;

    for (int i = tid; i < 16 * H; i += 256) {
        hL[i / H][i % H] = f2bf(0.f);
        cL[i / H][i % H] = 0.f;
    }
    __syncthreads();

    for (int t = 0; t < T; ++t) {
        v8f acc[NPW];
#pragma unroll
        for (int nt = 0; nt < NPW; ++nt)
            acc[nt] = v8f{0.f,0.f,0.f,0.f,0.f,0.f,0.f,0.f};

#pragma unroll
        for (int ks = 0; ks < KSTEPS; ++ks) {
            const int k0 = ks * 32;
            v8bf a0 = *(const v8bf*)(&hL[lm][k0 + 8 * hi]);
            v8bf a1 = *(const v8bf*)(&hL[lm][k0 + 16 + 8 * hi]);
            v16bf a = cat8(a0, a1);
#pragma unroll
            for (int nt = 0; nt < NPW; ++nt) {
                const int n = nbase + nt * 16 + lm;
                const __bf16* Wrow = Whh + (long)n * H + k0 + 8 * hi;
                v8bf b0 = *(const v8bf*)(Wrow);
                v8bf b1 = *(const v8bf*)(Wrow + 16);
                v16bf b = cat8(b0, b1);
                acc[nt] = __builtin_amdgcn_wmma_f32_16x16x32_bf16(
                    false, a, false, b, (short)0, acc[nt], false, false);
            }
        }
        // add precomputed input projection, apply gate activation, stash bf16
#pragma unroll
        for (int nt = 0; nt < NPW; ++nt) {
            const int ncol = nbase + nt * 16 + lm;
#pragma unroll
            for (int r = 0; r < 8; ++r) {
                const int m = r + 8 * hi;
                const long row = (long)btile * 16 + m;
                float g = acc[nt][r] + xg[(row * T + t) * N4 + ncol];
                float av = isTanh ? tanhf(g) : sigmoidf(g);
                gAct[m][ncol] = f2bf(av);
            }
        }
        if (t + 1 < T)   // pull next step's gate pre-activations toward L2/L0
            __builtin_prefetch(&xg[(((long)btile * 16 + lm) * T + (t + 1)) * N4 + nbase], 0, 1);
        __syncthreads();

        // c/h update: c = f*c + i*g ; h = o*tanh(c)
        for (int idx = tid; idx < 16 * H; idx += 256) {
            const int m = idx / H, n = idx % H;
            const float i = bf2f(gAct[m][n]);
            const float f = bf2f(gAct[m][n + H]);
            const float g = bf2f(gAct[m][n + 2 * H]);
            const float o = bf2f(gAct[m][n + 3 * H]);
            const float c = f * cL[m][n] + i * g;
            cL[m][n] = c;
            const float h = o * tanhf(c);
            hL[m][n] = f2bf(h);
            if constexpr (WRITE_SEQ) {
                outSeq[(((long)btile * 16 + m) * T + t) * H + n] = f2bf(h);
            } else {
                if (t == T - 1)
                    hLast[((long)btile * 16 + m) * H + n] = h;
            }
        }
        __syncthreads();
    }
}

// ------------------------- FC head -----------------------------------------

__global__ void k_head(const float* __restrict__ h, const float* __restrict__ Wfc,
                       const float* __restrict__ bfc, const float* __restrict__ Wout,
                       const float* __restrict__ bout, float* __restrict__ out) {
    const int b = threadIdx.x;
    if (b >= BATCH) return;
    float acc = 0.f;
    for (int j = 0; j < NFC1; ++j) {
        float s = bfc[j];
        for (int k = 0; k < H2; ++k)
            s += h[b * H2 + k] * Wfc[j * H2 + k];
        acc += fmaxf(s, 0.f) * Wout[j];
    }
    out[b] = acc + bout[0];
}

// ------------------------- launcher ----------------------------------------

extern "C" void kernel_launch(void* const* d_in, const int* in_sizes, int n_in,
                              void* d_out, int out_size, void* d_ws, size_t ws_size,
                              hipStream_t stream) {
    (void)in_sizes; (void)n_in; (void)out_size; (void)ws_size;
    const float* x     = (const float*)d_in[0];
    const float* Wih1  = (const float*)d_in[1];
    const float* Whh1  = (const float*)d_in[2];
    const float* bih1  = (const float*)d_in[3];
    const float* bhh1  = (const float*)d_in[4];
    const float* Wih2  = (const float*)d_in[5];
    const float* Whh2  = (const float*)d_in[6];
    const float* bih2  = (const float*)d_in[7];
    const float* bhh2  = (const float*)d_in[8];
    const float* W_fc1 = (const float*)d_in[9];
    const float* b_fc1 = (const float*)d_in[10];
    const float* W_out = (const float*)d_in[11];
    const float* b_out = (const float*)d_in[12];

    char* ws = (char*)d_ws;
    size_t off = 0;
    auto alloc = [&](size_t bytes) { size_t o = off; off = (off + bytes + 255) & ~(size_t)255; return o; };
    const size_t oW1  = alloc((size_t)4 * H1 * IN1 * 2);            // Wih1 bf16
    const size_t oU1  = alloc((size_t)4 * H1 * H1  * 2);            // Whh1 bf16
    const size_t oW2  = alloc((size_t)4 * H2 * H1  * 2);            // Wih2 bf16
    const size_t oU2  = alloc((size_t)4 * H2 * H2  * 2);            // Whh2 bf16
    const size_t oB1  = alloc((size_t)4 * H1 * 4);                  // bih1+bhh1
    const size_t oB2  = alloc((size_t)4 * H2 * 4);                  // bih2+bhh2
    const size_t oXB  = alloc((size_t)BATCH * SEQ * IN1 * 2);       // x bf16
    const size_t oXG1 = alloc((size_t)BATCH * SEQ * 4 * H1 * 4);    // gate preacts L1 (f32)
    const size_t oO1  = alloc((size_t)BATCH * SEQ * H1 * 2);        // h sequence L1 (bf16)
    const size_t oXG2 = alloc((size_t)BATCH * SEQ * 4 * H2 * 4);    // gate preacts L2 (f32)
    const size_t oH2  = alloc((size_t)BATCH * H2 * 4);              // final h L2 (f32)

    // weight/bias/x prep (bandwidth-bound, deterministic, redone every call)
    k_f2bf<<<4096, 256, 0, stream>>>(x, (__bf16*)(ws + oXB), BATCH * SEQ * IN1);
    k_f2bf<<<512, 256, 0, stream>>>(Wih1, (__bf16*)(ws + oW1), 4 * H1 * IN1);
    k_f2bf<<<256, 256, 0, stream>>>(Whh1, (__bf16*)(ws + oU1), 4 * H1 * H1);
    k_f2bf<<<256, 256, 0, stream>>>(Wih2, (__bf16*)(ws + oW2), 4 * H2 * H1);
    k_f2bf<<<128, 256, 0, stream>>>(Whh2, (__bf16*)(ws + oU2), 4 * H2 * H2);
    k_addbias<<<4, 256, 0, stream>>>(bih1, bhh1, (float*)(ws + oB1), 4 * H1);
    k_addbias<<<2, 256, 0, stream>>>(bih2, bhh2, (float*)(ws + oB2), 4 * H2);

    // layer 1: xg1 = x @ Wih1^T + (bih1+bhh1)   (M=32768, K=2048, N=1024)
    k_xproj<8><<<BATCH * SEQ / 32, 256, 0, stream>>>(
        (const __bf16*)(ws + oXB), (const __bf16*)(ws + oW1),
        (const float*)(ws + oB1), (float*)(ws + oXG1), IN1, 4 * H1);
    // layer 1 recurrence (4 blocks of 16 batch rows)
    k_lstm_rec<H1, true><<<BATCH / 16, 256, 0, stream>>>(
        (const float*)(ws + oXG1), (const __bf16*)(ws + oU1),
        (__bf16*)(ws + oO1), nullptr, SEQ);
    // layer 2: xg2 = h1 @ Wih2^T + (bih2+bhh2)  (M=32768, K=256, N=512)
    k_xproj<4><<<BATCH * SEQ / 32, 256, 0, stream>>>(
        (const __bf16*)(ws + oO1), (const __bf16*)(ws + oW2),
        (const float*)(ws + oB2), (float*)(ws + oXG2), H1, 4 * H2);
    // layer 2 recurrence -> only final h needed
    k_lstm_rec<H2, false><<<BATCH / 16, 256, 0, stream>>>(
        (const float*)(ws + oXG2), (const __bf16*)(ws + oU2),
        nullptr, (float*)(ws + oH2), SEQ);
    // FC head
    k_head<<<1, 64, 0, stream>>>((const float*)(ws + oH2), W_fc1, b_fc1,
                                 W_out, b_out, (float*)d_out);
}
